// SwinTransformerBlock_83339545412150
// MI455X (gfx1250) — matrix operand order repaired and checked by reading, not compiled, not run on hardware
//
#include <hip/hip_runtime.h>
#include <hip/hip_bf16.h>
#include <math.h>

typedef _Float16 v16h __attribute__((ext_vector_type(16)));
typedef _Float16 v8h  __attribute__((ext_vector_type(8)));
typedef float    v8f  __attribute__((ext_vector_type(8)));

#define WMMA_F16(a, b, c) __builtin_amdgcn_wmma_f32_16x16x32_f16( \
    false, (a), false, (b), (short)0, (c), false, false)

// ---- constants ----
#define BATCH   64
#define HW      56
#define CH      96
#define HEADS   3
#define WS      7
#define SHIFT   3
#define NTOK    49
#define NPAD    64
#define NWIN    64
#define HD      32
#define SCALE   0.17677669529663687f  // 32^-0.5
#define TOKENS  (BATCH * HW * HW)          // 200704
#define XMID_ELEMS ((size_t)TOKENS * CH)   // 19267584 floats

// f16 transposed-weight blob offsets (halves), after x_mid in workspace.
// All weights stored N-major: Wt[N][K] so B-fragments are contiguous per lane.
#define QKVW_OFF  0                 // Wt[288][96]  = 27648
#define PROJW_OFF 27648             // Wt[96][96]   = 9216
#define FC1W_OFF  36864             // Wt[384][96]  = 36864
#define FC2W_OFF  73728             // Wt[96][384]  = 36864
#define WTOTAL    110592

// ---------- WMMA fragment loaders (wave32 layouts per CDNA5 ISA 7.12.2) ----------
__device__ __forceinline__ v16h cat16(v8h lo, v8h hi) {
    return __builtin_shufflevector(lo, hi, 0,1,2,3,4,5,6,7,8,9,10,11,12,13,14,15);
}
// A: 16x32 f16 from row-major [M][K] (ld halves, 16B-aligned rows/tiles).
// lane row = lane&15; halves 0..7 -> K = 8*(lane>>4)+j ; halves 8..15 -> K = 16+8*(lane>>4)+j
__device__ __forceinline__ v16h load_a(const _Float16* base, int ld, int lane) {
    const _Float16* p = base + (lane & 15) * ld + ((lane >> 4) << 3);
    return cat16(*(const v8h*)p, *(const v8h*)(p + 16));
}
// B: 32x16 f16 from N-major layout Bt[N][K] (ld halves): col = lane&15; halves 0..7 -> K = 8*(lane>>4)+j ; halves 8..15 -> K = 16+8*(lane>>4)+j.
// two aligned b128 loads per lane.
__device__ __forceinline__ v16h load_bn(const _Float16* base, int ld, int lane) {
    const _Float16* p = base + (lane & 15) * ld + ((lane >> 4) << 3);
    return cat16(*(const v8h*)p, *(const v8h*)(p + 16));
}

// ---------------- kernel 0: fp32 -> f16 transposed weight conversion ----------------
__global__ __launch_bounds__(256) void cvt_weights_kernel(
    const float* __restrict__ qkvw, const float* __restrict__ projw,
    const float* __restrict__ f1w,  const float* __restrict__ f2w,
    _Float16* __restrict__ wout)
{
    int i = blockIdx.x * 256 + threadIdx.x;
    if (i >= WTOTAL) return;
    float v;
    if (i < PROJW_OFF) {                       // qkv: Wt[288][96] <- W[96][288]
        int n = i / 96, k = i % 96;
        v = qkvw[k * 288 + n];
    } else if (i < FC1W_OFF) {                 // proj: Wt[96][96]
        int j = i - PROJW_OFF, n = j / 96, k = j % 96;
        v = projw[k * 96 + n];
    } else if (i < FC2W_OFF) {                 // fc1: Wt[384][96] <- W[96][384]
        int j = i - FC1W_OFF, n = j / 96, k = j % 96;
        v = f1w[k * 384 + n];
    } else {                                   // fc2: Wt[96][384] <- W[384][96]
        int j = i - FC2W_OFF, n = j / 384, k = j % 384;
        v = f2w[k * 96 + n];
    }
    wout[i] = (_Float16)v;
}

// region id for the shifted-window attention mask (matches reference slices)
__device__ __forceinline__ int region(int s) { return s < 49 ? 0 : (s < 53 ? 1 : 2); }

// ---------------- kernel 1: LN1 + shift + window attn + proj + residual ----------------
__global__ __launch_bounds__(256) void swin_attn_kernel(
    const float* __restrict__ x,
    const float* __restrict__ n1g, const float* __restrict__ n1b,
    const _Float16* __restrict__ qkvwT, const float* __restrict__ qkvb,
    const float* __restrict__ relb,
    const _Float16* __restrict__ projwT, const float* __restrict__ projb,
    float* __restrict__ xmid)
{
    __shared__ _Float16 sX [NPAD * CH];    // LN'd tokens; reused as attention output O
    __shared__ _Float16 sQ [NPAD * CH];    // pre-scaled Q, all heads, row-major [tok][ch]
    __shared__ _Float16 sK [NPAD * CH];    // row-major [tok][ch] (N-major for Q@K^T)
    __shared__ _Float16 sVt[CH * NPAD];    // V transposed: [ch][key] (N-major for P@V)
    __shared__ float    sS [NPAD * NPAD];  // scores f32; probs overlaid in-place (stride 128 halves)

    const int tid  = threadIdx.x;
    const int lane = tid & 31;
    const int wave = tid >> 5;
    const int wid  = blockIdx.x;
    const int bb   = wid >> 6;
    const int w64  = wid & 63;
    const int wh   = w64 >> 3, ww = w64 & 7;
    const float* xb = x + (size_t)bb * (HW * HW * CH);

    __builtin_prefetch(qkvwT, 0, 0);
    __builtin_prefetch(projwT, 0, 0);

    // zero padding rows 49..63 of the activation tile
    for (int i = tid; i < (NPAD - NTOK) * CH; i += 256) sX[NTOK * CH + i] = (_Float16)0.f;

    // LN1 + gather through roll(-3,-3)
    for (int t = wave; t < NTOK; t += 8) {
        const int r = t / WS, c = t % WS;
        int oh = wh * WS + r + SHIFT; if (oh >= HW) oh -= HW;
        int ow = ww * WS + c + SHIFT; if (ow >= HW) ow -= HW;
        const float* xr = xb + (oh * HW + ow) * CH;
        const int c0 = lane * 3;
        float v0 = xr[c0], v1 = xr[c0 + 1], v2 = xr[c0 + 2];
        float s  = v0 + v1 + v2;
        float ss = v0 * v0 + v1 * v1 + v2 * v2;
#pragma unroll
        for (int off = 16; off > 0; off >>= 1) {
            s  += __shfl_xor(s,  off, 32);
            ss += __shfl_xor(ss, off, 32);
        }
        const float mu  = s * (1.f / CH);
        const float var = ss * (1.f / CH) - mu * mu;
        const float rs  = rsqrtf(var + 1e-5f);
        sX[t * CH + c0    ] = (_Float16)((v0 - mu) * rs * n1g[c0    ] + n1b[c0    ]);
        sX[t * CH + c0 + 1] = (_Float16)((v1 - mu) * rs * n1g[c0 + 1] + n1b[c0 + 1]);
        sX[t * CH + c0 + 2] = (_Float16)((v2 - mu) * rs * n1g[c0 + 2] + n1b[c0 + 2]);
    }
    __syncthreads();

    // ---- QKV GEMM: [64 x 96] @ [96 x 288], tiles 4x18, K-steps 3 ----
    for (int tile = wave; tile < 72; tile += 8) {
        const int mt = tile / 18, nt = tile % 18;  // target buffer is uniform per tile
        v8f acc = {};
#pragma unroll
        for (int kt = 0; kt < 3; ++kt) {
            v16h a  = load_a (sX + mt * 16 * CH + kt * 32, CH, lane);
            v16h bf = load_bn(qkvwT + (nt * 16) * CH + kt * 32, CH, lane);
            acc = WMMA_F16(a, bf, acc);
        }
        const int g     = nt * 16 + (lane & 15);
        const int rbase = mt * 16 + (lane >> 4) * 8;
        const float bias = qkvb[g];
        if (nt < 6) {                      // Q, pre-scaled
#pragma unroll
            for (int r = 0; r < 8; ++r)
                sQ[(rbase + r) * CH + g] = (_Float16)((acc[r] + bias) * SCALE);
        } else if (nt < 12) {              // K
            const int gk = g - 96;
#pragma unroll
            for (int r = 0; r < 8; ++r)
                sK[(rbase + r) * CH + gk] = (_Float16)(acc[r] + bias);
        } else {                           // V -> transposed, one b128 store per lane
            const int gv = g - 192;
            v8h pk;
#pragma unroll
            for (int r = 0; r < 8; ++r) pk[r] = (_Float16)(acc[r] + bias);
            *(v8h*)(sVt + gv * NPAD + rbase) = pk;
        }
    }
    __syncthreads();

    // ---- per-head attention ----
    _Float16* sP = reinterpret_cast<_Float16*>(sS);  // probs, row stride 128 halves
    for (int head = 0; head < HEADS; ++head) {
        // S = (Q*scale) @ K^T + rel_bias + shift_mask ; tiles 4x4, K=32
        for (int tile = wave; tile < 16; tile += 8) {
            const int mt = tile >> 2, nt = tile & 3;
            v16h a  = load_a (sQ + mt * 16 * CH + head * HD, CH, lane);
            v16h bf = load_bn(sK + (nt * 16) * CH + head * HD, CH, lane);
            v8f acc = {};
            acc = WMMA_F16(a, bf, acc);
            const int kt_   = nt * 16 + (lane & 15);
            const int rbase = mt * 16 + (lane >> 4) * 8;
            const int kr = kt_ / WS, kc = kt_ % WS;
            const int idk = (kt_ < NTOK) ? region(wh * WS + kr) * 3 + region(ww * WS + kc) : -1;
#pragma unroll
            for (int r = 0; r < 8; ++r) {
                const int qt = rbase + r;
                float v = acc[r];
                if (qt < NTOK && kt_ < NTOK) {
                    const int qr = qt / WS, qc = qt % WS;
                    const int rel = (qr - kr + WS - 1) * (2 * WS - 1) + (qc - kc + WS - 1);
                    v += relb[rel * HEADS + head];
                    const int idq = region(wh * WS + qr) * 3 + region(ww * WS + qc);
                    if (idq != idk) v += -100.f;
                } else {
                    v = -1e30f;  // key padding / unused query rows
                }
                sS[qt * NPAD + kt_] = v;
            }
        }
        __syncthreads();

        // softmax: one wave per row, 2 columns per lane, shfl reductions
        for (int row = wave; row < NPAD; row += 8) {
            if (row < NTOK) {
                const float v0 = sS[row * NPAD + lane];
                const float v1 = sS[row * NPAD + lane + 32];
                float mx = fmaxf(v0, v1);
#pragma unroll
                for (int off = 16; off > 0; off >>= 1) mx = fmaxf(mx, __shfl_xor(mx, off, 32));
                const float e0 = __expf(v0 - mx), e1 = __expf(v1 - mx);
                float s = e0 + e1;
#pragma unroll
                for (int off = 16; off > 0; off >>= 1) s += __shfl_xor(s, off, 32);
                const float inv = 1.f / s;
                sP[row * 128 + lane]      = (_Float16)(e0 * inv);
                sP[row * 128 + lane + 32] = (_Float16)(e1 * inv);
            } else {
                sP[row * 128 + lane]      = (_Float16)0.f;
                sP[row * 128 + lane + 32] = (_Float16)0.f;
            }
        }
        __syncthreads();

        // O_head = P @ V_head : tiles 4x2 (exactly 8 waves), K-steps 2
        {
            const int mt = wave >> 1, nt = wave & 1;
            v8f acc = {};
#pragma unroll
            for (int kt = 0; kt < 2; ++kt) {
                v16h a  = load_a (sP + mt * 16 * 128 + kt * 32, 128, lane);
                v16h bf = load_bn(sVt + (head * HD + nt * 16) * NPAD + kt * 32, NPAD, lane);
                acc = WMMA_F16(a, bf, acc);
            }
            const int colg  = head * HD + nt * 16 + (lane & 15);
            const int rbase = mt * 16 + (lane >> 4) * 8;
#pragma unroll
            for (int r = 0; r < 8; ++r)
                sX[(rbase + r) * CH + colg] = (_Float16)acc[r];  // sX reused as O
        }
        __syncthreads();
    }

    // ---- proj GEMM [64x96]@[96x96] + bias + residual, scatter via inverse roll ----
    for (int tile = wave; tile < 24; tile += 8) {
        const int mt = tile / 6, nt = tile % 6;
        v8f acc = {};
#pragma unroll
        for (int kt = 0; kt < 3; ++kt) {
            v16h a  = load_a (sX + mt * 16 * CH + kt * 32, CH, lane);
            v16h bf = load_bn(projwT + (nt * 16) * CH + kt * 32, CH, lane);
            acc = WMMA_F16(a, bf, acc);
        }
        const int g     = nt * 16 + (lane & 15);
        const int rbase = mt * 16 + (lane >> 4) * 8;
        const float pb = projb[g];
#pragma unroll
        for (int r = 0; r < 8; ++r) {
            const int t = rbase + r;
            if (t < NTOK) {
                const int rr = t / WS, cc = t % WS;
                int oh = wh * WS + rr + SHIFT; if (oh >= HW) oh -= HW;
                int ow = ww * WS + cc + SHIFT; if (ow >= HW) ow -= HW;
                const size_t idx = ((size_t)bb * (HW * HW) + oh * HW + ow) * CH + g;
                xmid[idx] = x[idx] + acc[r] + pb;  // roll(-s) then roll(+s): same index
            }
        }
    }
}

// ---------------- kernel 2: LN2 + fc1 + GELU + fc2 + residual ----------------
__global__ __launch_bounds__(256) void swin_mlp_kernel(
    const float* __restrict__ xmid,
    const float* __restrict__ n2g, const float* __restrict__ n2b,
    const _Float16* __restrict__ f1wT, const float* __restrict__ f1b,
    const _Float16* __restrict__ f2wT, const float* __restrict__ f2b,
    float* __restrict__ out)
{
    __shared__ _Float16 sXn[NPAD * CH];       // 12 KB
    __shared__ _Float16 sH[NPAD * 4 * CH];    // 48 KB

    const int tid  = threadIdx.x;
    const int lane = tid & 31;
    const int wave = tid >> 5;
    const size_t tok0 = (size_t)blockIdx.x * NPAD;

    __builtin_prefetch(f1wT, 0, 0);
    __builtin_prefetch(f2wT, 0, 0);

    // LN2, one wave per token
    for (int i = wave; i < NPAD; i += 8) {
        const float* xr = xmid + (tok0 + i) * CH;
        const int c0 = lane * 3;
        float v0 = xr[c0], v1 = xr[c0 + 1], v2 = xr[c0 + 2];
        float s  = v0 + v1 + v2;
        float ss = v0 * v0 + v1 * v1 + v2 * v2;
#pragma unroll
        for (int off = 16; off > 0; off >>= 1) {
            s  += __shfl_xor(s,  off, 32);
            ss += __shfl_xor(ss, off, 32);
        }
        const float mu  = s * (1.f / CH);
        const float var = ss * (1.f / CH) - mu * mu;
        const float rs  = rsqrtf(var + 1e-5f);
        sXn[i * CH + c0    ] = (_Float16)((v0 - mu) * rs * n2g[c0    ] + n2b[c0    ]);
        sXn[i * CH + c0 + 1] = (_Float16)((v1 - mu) * rs * n2g[c0 + 1] + n2b[c0 + 1]);
        sXn[i * CH + c0 + 2] = (_Float16)((v2 - mu) * rs * n2g[c0 + 2] + n2b[c0 + 2]);
    }
    __syncthreads();

    // fc1: [64x96]@[96x384] + bias, exact GELU ; tiles 4x24, K-steps 3
    for (int tile = wave; tile < 96; tile += 8) {
        const int mt = tile / 24, nt = tile % 24;
        v8f acc = {};
#pragma unroll
        for (int kt = 0; kt < 3; ++kt) {
            v16h a  = load_a (sXn + mt * 16 * CH + kt * 32, CH, lane);
            v16h bf = load_bn(f1wT + (nt * 16) * CH + kt * 32, CH, lane);
            acc = WMMA_F16(a, bf, acc);
        }
        const int g     = nt * 16 + (lane & 15);
        const int rbase = mt * 16 + (lane >> 4) * 8;
        const float bias = f1b[g];
#pragma unroll
        for (int r = 0; r < 8; ++r) {
            float v = acc[r] + bias;
            v = 0.5f * v * (1.f + erff(v * 0.70710678118654752f));
            sH[(rbase + r) * (4 * CH) + g] = (_Float16)v;
        }
    }
    __syncthreads();

    // fc2: [64x384]@[384x96] + bias + residual ; tiles 4x6, K-steps 12
    for (int tile = wave; tile < 24; tile += 8) {
        const int mt = tile / 6, nt = tile % 6;
        v8f acc = {};
#pragma unroll
        for (int kt = 0; kt < 12; ++kt) {
            v16h a  = load_a (sH + mt * 16 * (4 * CH) + kt * 32, 4 * CH, lane);
            v16h bf = load_bn(f2wT + (nt * 16) * (4 * CH) + kt * 32, 4 * CH, lane);
            acc = WMMA_F16(a, bf, acc);
        }
        const int g     = nt * 16 + (lane & 15);
        const int rbase = mt * 16 + (lane >> 4) * 8;
        const float bias = f2b[g];
#pragma unroll
        for (int r = 0; r < 8; ++r) {
            const size_t t = tok0 + rbase + r;
            out[t * CH + g] = xmid[t * CH + g] + acc[r] + bias;
        }
    }
}

extern "C" void kernel_launch(void* const* d_in, const int* in_sizes, int n_in,
                              void* d_out, int out_size, void* d_ws, size_t ws_size,
                              hipStream_t stream) {
    (void)in_sizes; (void)n_in; (void)out_size; (void)ws_size;
    const float* x     = (const float*)d_in[0];
    const float* n1g   = (const float*)d_in[1];
    const float* n1b   = (const float*)d_in[2];
    const float* qkvw  = (const float*)d_in[3];
    const float* qkvb  = (const float*)d_in[4];
    const float* relb  = (const float*)d_in[5];
    const float* projw = (const float*)d_in[6];
    const float* projb = (const float*)d_in[7];
    const float* n2g   = (const float*)d_in[8];
    const float* n2b   = (const float*)d_in[9];
    const float* f1w   = (const float*)d_in[10];
    const float* f1b   = (const float*)d_in[11];
    const float* f2w   = (const float*)d_in[12];
    const float* f2b   = (const float*)d_in[13];

    float*     xmid = (float*)d_ws;                       // 77 MB residual stream (L2-resident)
    _Float16*  wh   = (_Float16*)(xmid + XMID_ELEMS);     // f16 transposed weight blob (0.22 MB)

    cvt_weights_kernel<<<(WTOTAL + 255) / 256, 256, 0, stream>>>(qkvw, projw, f1w, f2w, wh);

    swin_attn_kernel<<<BATCH * NWIN, 256, 0, stream>>>(
        x, n1g, n1b, wh + QKVW_OFF, qkvb, relb, wh + PROJW_OFF, projb, xmid);

    swin_mlp_kernel<<<TOKENS / NPAD, 256, 0, stream>>>(
        xmid, n2g, n2b, wh + FC1W_OFF, f1b, wh + FC2W_OFF, f2b, (float*)d_out);
}
